// MeshDeformationBlock_28269474742810
// MI455X (gfx1250) — matrix operand order repaired and checked by reading, not compiled, not run on hardware
//
#include <hip/hip_runtime.h>

typedef __bf16 bf16_t;
typedef __attribute__((ext_vector_type(16))) __bf16 v16bf;
typedef __attribute__((ext_vector_type(8)))  __bf16 v8bf;
typedef __attribute__((ext_vector_type(8)))  float  v8f;

#define N_NODES 8192
#define HID     192
#define SIDE    64
#define K_FULL  960

// ---------------------------------------------------------------------------
// WMMA helper: D = A(16x32 bf16) * B(32x16 bf16) + C(16x16 f32)
// ---------------------------------------------------------------------------
__device__ __forceinline__ v8f wmma_bf16(v16bf a, v16bf b, v8f c) {
  return __builtin_amdgcn_wmma_f32_16x16x32_bf16(
      /*neg_a=*/false, a, /*neg_b=*/false, b,
      /*c_mod=*/(short)0, c, /*reuse_a=*/false, /*reuse_b=*/false);
}

// A-fragment (16x32, row-major source, row stride ld).
// Layout: lane L holds row m = L%16; its 16 bf16 are the contiguous
// K-run starting at (L/16)*16 (matches the B fragments).
__device__ __forceinline__ v16bf load_a_frag(const bf16_t* base, int ld,
                                             int row0, int k0, int lane) {
  int m  = row0 + (lane & 15);
  int kb = k0 + ((lane >> 4) << 4);
  const bf16_t* p = base + (size_t)m * ld + kb;
  return *(const v16bf*)p;            // K = kb .. kb+15
}

// B-fragment (32x16) from a column-major store (column stride ldK).
// Layout (same k order as A): lane L holds column n = L%16; its 16 bf16 are the contiguous
// K-run starting at (L/16)*16 -> single 32-byte load.
__device__ __forceinline__ v16bf load_b_frag_cm(const bf16_t* base, int ldK,
                                                int n0, int k0, int lane) {
  int n  = n0 + (lane & 15);
  int kb = k0 + ((lane >> 4) << 4);
  return *(const v16bf*)(base + (size_t)n * ldK + kb);
}

// ---------------------------------------------------------------------------
// prep_adj: adj fp32 -> bf16, plus inv_norm[row] = 1 / rowsum(adj)
// ---------------------------------------------------------------------------
__global__ void prep_adj(const float* __restrict__ adj, bf16_t* __restrict__ adjB,
                         float* __restrict__ inv_norm) {
  int row = blockIdx.x;
  const float* a = adj + (size_t)row * N_NODES;
  bf16_t* o = adjB + (size_t)row * N_NODES;
  float s = 0.f;
  for (int c = threadIdx.x; c < N_NODES; c += 256) {
    float v = a[c];
    s += v;
    o[c] = (bf16_t)v;
  }
  __shared__ float red[256];
  red[threadIdx.x] = s;
  __syncthreads();
  for (int off = 128; off > 0; off >>= 1) {
    if (threadIdx.x < off) red[threadIdx.x] += red[threadIdx.x + off];
    __syncthreads();
  }
  if (threadIdx.x == 0) inv_norm[row] = 1.0f / red[0];
}

// full = concat(features, pooled) -> bf16 [8192,960]
__global__ void pack_full(const float* __restrict__ feat,
                          const float* __restrict__ pooled,
                          bf16_t* __restrict__ fullB) {
  int i = blockIdx.x * 256 + threadIdx.x;
  if (i >= N_NODES * K_FULL) return;
  int r = i / K_FULL, c = i % K_FULL;
  float v = (c < HID) ? feat[(size_t)r * HID + c]
                      : pooled[(size_t)r * 768 + (c - HID)];
  fullB[i] = (bf16_t)v;
}

// W fp32 row-major [K,OUT] -> bf16 col-major [OUT][K]
__global__ void convert_w(const float* __restrict__ W, bf16_t* __restrict__ Wt,
                          int K, int OUT) {
  int i = blockIdx.x * 256 + threadIdx.x;
  if (i >= K * OUT) return;
  int n = i / K, k = i % K;
  Wt[i] = (bf16_t)W[(size_t)k * OUT + n];
}

// support[8192,192] f32 = A_bf16[8192,K] @ Wt(col-major [192][K])
// block = 256 thr = 8 waves; block tile = 32 rows x 192 cols;
// wave w: m-tile = w>>2, n-cols = (w&3)*48 + 3 fragments. grid = 256 blocks.
__global__ void gemm_xw(const bf16_t* __restrict__ A, const bf16_t* __restrict__ Wt,
                        float* __restrict__ C, int K) {
  int wave = threadIdx.x >> 5, lane = threadIdx.x & 31;
  int row0  = blockIdx.x * 32 + (wave >> 2) * 16;
  int ncol0 = (wave & 3) * 48;
  v8f acc[3] = {};
  for (int k = 0; k < K; k += 32) {
    v16bf a = load_a_frag(A, K, row0, k, lane);
#pragma unroll
    for (int t = 0; t < 3; ++t) {
      v16bf b = load_b_frag_cm(Wt, K, ncol0 + 16 * t, k, lane);
      acc[t] = wmma_bf16(a, b, acc[t]);
    }
  }
#pragma unroll
  for (int t = 0; t < 3; ++t) {
#pragma unroll
    for (int i = 0; i < 8; ++i) {
      int r = row0 + i + ((lane >> 4) << 3);
      int c = ncol0 + 16 * t + (lane & 15);
      C[(size_t)r * HID + c] = acc[t][i];
    }
  }
}

// From support: pack side channels scaled by inv_norm into column-major bf16
// sB[64][8192]; local channels (>=64) get bias+ReLU into xf (f32) + xbf (bf16).
__global__ void post_support(const float* __restrict__ sup,
                             const float* __restrict__ inv_norm,
                             const float* __restrict__ bias,
                             bf16_t* __restrict__ sB,
                             float* __restrict__ xf, bf16_t* __restrict__ xbf) {
  int i = blockIdx.x * 256 + threadIdx.x;
  if (i >= N_NODES * HID) return;
  int j = i / HID, n = i % HID;
  float v = sup[i];
  if (n < SIDE) {
    sB[(size_t)n * N_NODES + j] = (bf16_t)(v * inv_norm[j]);
  } else {
    float o = v + bias[n];
    o = o > 0.f ? o : 0.f;
    xf[i]  = o;
    xbf[i] = (bf16_t)o;
  }
}

// ---------------------------------------------------------------------------
// side1 = adjB[8192,8192] @ sB(col-major [64][8192]); fused bias+ReLU into
// xf/xbf columns 0..63.
// block = 8 waves; wave tile = 32 rows x 64 cols (2 M-tiles x 4 N-tiles,
// 8 f32 accumulators). block covers 256 rows; grid = 32.
// The shared 32x64 bf16 B tile (4 KB) is staged into LDS with
// global_load_async_to_lds_b128 (ASYNCcnt-tracked, double buffered) so the
// copy of tile i+1 overlaps the WMMAs of tile i and 8 waves share one copy.
// ---------------------------------------------------------------------------
__global__ void adj_side(const bf16_t* __restrict__ adjB,
                         const bf16_t* __restrict__ sB,
                         const float* __restrict__ bias,
                         float* __restrict__ xf, bf16_t* __restrict__ xbf) {
  // ldsB[buf][n*32 + k] : column-major tile, K-run contiguous per column.
  __shared__ __align__(32) bf16_t ldsB[2][32 * 64];

  int wave = threadIdx.x >> 5, lane = threadIdx.x & 31;
  int row0 = blockIdx.x * 256 + wave * 32;

  // async stage of one 32(K) x 64(N) tile: thread t copies 16 bytes of
  // column n = t/4, chunk c = t%4.
  int sn = threadIdx.x >> 2;
  int sc4 = threadIdx.x & 3;
  unsigned lds_base0 = (unsigned)(uintptr_t)(&ldsB[0][0]);
  unsigned lds_base1 = (unsigned)(uintptr_t)(&ldsB[1][0]);
  unsigned lds_elem = (unsigned)(sn * 32 + sc4 * 8) * 2u;  // byte offset in tile
  const bf16_t* gcol = sB + (size_t)sn * N_NODES + sc4 * 8;

  {
    unsigned lds_addr = lds_base0 + lds_elem;
    unsigned long long ga = (unsigned long long)(uintptr_t)gcol;  // k0 = 0
    asm volatile("global_load_async_to_lds_b128 %0, %1, off"
                 :: "v"(lds_addr), "v"(ga) : "memory");
  }

  v8f acc[2][4] = {};
  const int NK = N_NODES / 32;  // 256 K-tiles
  for (int i = 0; i < NK; ++i) {
    // stage next tile into the other buffer (last iter re-stages k=0, unused)
    {
      int knext = (i + 1 < NK) ? (i + 1) * 32 : 0;
      unsigned lds_addr = ((i & 1) ? lds_base0 : lds_base1) + lds_elem;
      unsigned long long ga = (unsigned long long)(uintptr_t)(gcol + knext);
      asm volatile("global_load_async_to_lds_b128 %0, %1, off"
                   :: "v"(lds_addr), "v"(ga) : "memory");
    }
    // retire the tile for this iteration (one older op outstanding max)
    asm volatile("s_wait_asynccnt 0x1" ::: "memory");
    __syncthreads();

    const bf16_t* bt = (i & 1) ? &ldsB[1][0] : &ldsB[0][0];
    v16bf bfrag[4];
#pragma unroll
    for (int t = 0; t < 4; ++t) {
      int n = 16 * t + (lane & 15);
      int kb = (lane >> 4) << 4;
      bfrag[t] = *(const v16bf*)(bt + n * 32 + kb);
    }
#pragma unroll
    for (int mt = 0; mt < 2; ++mt) {
      v16bf a = load_a_frag(adjB, N_NODES, row0 + 16 * mt, i * 32, lane);
#pragma unroll
      for (int t = 0; t < 4; ++t) acc[mt][t] = wmma_bf16(a, bfrag[t], acc[mt][t]);
    }
    __syncthreads();  // all waves done reading before next overwrite
  }

#pragma unroll
  for (int mt = 0; mt < 2; ++mt) {
#pragma unroll
    for (int t = 0; t < 4; ++t) {
#pragma unroll
      for (int i = 0; i < 8; ++i) {
        int r = row0 + 16 * mt + i + ((lane >> 4) << 3);
        int c = 16 * t + (lane & 15);
        float o = acc[mt][t][i] + bias[c];
        o = o > 0.f ? o : 0.f;
        xf[(size_t)r * HID + c]  = o;
        xbf[(size_t)r * HID + c] = (bf16_t)o;
      }
    }
  }
}

// feats = (a + x)/2 ; also emit bf16 copy for next layer's GEMM input.
__global__ void avg_kernel(const float* __restrict__ a, const float* __restrict__ x,
                           float* __restrict__ out, bf16_t* __restrict__ outB) {
  int i = blockIdx.x * 256 + threadIdx.x;
  if (i >= N_NODES * HID) return;
  float v = (a[i] + x[i]) * 0.5f;
  out[i]  = v;
  outB[i] = (bf16_t)v;
}

// gc15 support: s3[j,c] = feats[j,:] . W15[:,c]; also scaled bf16 for c<2.
__global__ void coords_support(const float* __restrict__ feats,
                               const float* __restrict__ W15,
                               const float* __restrict__ inv_norm,
                               float* __restrict__ s3, bf16_t* __restrict__ sc) {
  int i = blockIdx.x * 256 + threadIdx.x;
  if (i >= N_NODES * 3) return;
  int j = i / 3, c = i % 3;
  float acc = 0.f;
  for (int k = 0; k < HID; ++k) acc += feats[(size_t)j * HID + k] * W15[k * 3 + c];
  s3[i] = acc;
  if (c < 2) sc[(size_t)c * N_NODES + j] = (bf16_t)(acc * inv_norm[j]);
}

// coords[i,0:2] = adj row i . sc[0:2] + b ; coords[i,2] = s3[i,2] + b[2].
__global__ void coords_side(const bf16_t* __restrict__ adjB,
                            const bf16_t* __restrict__ sc,
                            const float* __restrict__ s3,
                            const float* __restrict__ b15,
                            float* __restrict__ coords) {
  int row = blockIdx.x;
  const bf16_t* a = adjB + (size_t)row * N_NODES;
  float s0 = 0.f, s1 = 0.f;
  for (int j = threadIdx.x; j < N_NODES; j += 256) {
    float av = (float)a[j];
    s0 += av * (float)sc[j];
    s1 += av * (float)sc[N_NODES + j];
  }
  __shared__ float r0[256], r1[256];
  r0[threadIdx.x] = s0; r1[threadIdx.x] = s1;
  __syncthreads();
  for (int off = 128; off > 0; off >>= 1) {
    if (threadIdx.x < off) {
      r0[threadIdx.x] += r0[threadIdx.x + off];
      r1[threadIdx.x] += r1[threadIdx.x + off];
    }
    __syncthreads();
  }
  if (threadIdx.x == 0) {
    coords[(size_t)row * 3 + 0] = r0[0] + b15[0];
    coords[(size_t)row * 3 + 1] = r1[0] + b15[1];
    coords[(size_t)row * 3 + 2] = s3[(size_t)row * 3 + 2] + b15[2];
  }
}

// ---------------------------------------------------------------------------
extern "C" void kernel_launch(void* const* d_in, const int* in_sizes, int n_in,
                              void* d_out, int out_size, void* d_ws, size_t ws_size,
                              hipStream_t stream) {
  (void)in_sizes; (void)n_in; (void)out_size; (void)ws_size;
  const float* features = (const float*)d_in[0];   // [8192,192]
  const float* pooled   = (const float*)d_in[1];   // [8192,768]
  const float* adj      = (const float*)d_in[2];   // [8192,8192]
  const float* W1       = (const float*)d_in[3];   // [960,192]
  const float* b1       = (const float*)d_in[4];   // [192]
  const float* Wm       = (const float*)d_in[5];   // [12,192,192]
  const float* bm       = (const float*)d_in[6];   // [12,192]
  const float* W15      = (const float*)d_in[7];   // [192,3]
  const float* b15      = (const float*)d_in[8];   // [3]

  float* feats  = (float*)d_out;                        // [8192,192]
  float* coords = feats + (size_t)N_NODES * HID;        // [8192,3]

  // workspace carve-up (256B aligned)
  char* wp = (char*)d_ws;
  auto carve = [&](size_t bytes) {
    char* p = wp;
    wp += (bytes + 255) & ~(size_t)255;
    return p;
  };
  bf16_t* adjB     = (bf16_t*)carve((size_t)N_NODES * N_NODES * sizeof(bf16_t)); // 128 MB
  float*  inv_norm = (float*) carve((size_t)N_NODES * sizeof(float));
  bf16_t* fullB    = (bf16_t*)carve((size_t)N_NODES * K_FULL * sizeof(bf16_t));
  bf16_t* Wbf      = (bf16_t*)carve((size_t)K_FULL * HID * sizeof(bf16_t));
  float*  support  = (float*) carve((size_t)N_NODES * HID * sizeof(float));
  bf16_t* sB       = (bf16_t*)carve((size_t)SIDE * N_NODES * sizeof(bf16_t));
  float*  xf       = (float*) carve((size_t)N_NODES * HID * sizeof(float));
  bf16_t* xbf      = (bf16_t*)carve((size_t)N_NODES * HID * sizeof(bf16_t));
  bf16_t* featsB   = (bf16_t*)carve((size_t)N_NODES * HID * sizeof(bf16_t));
  float*  s3       = (float*) carve((size_t)N_NODES * 3 * sizeof(float));
  bf16_t* sc       = (bf16_t*)carve((size_t)2 * N_NODES * sizeof(bf16_t));

  const int EW_GRID = (N_NODES * HID) / 256;  // 6144

  auto run_layer = [&](const bf16_t* xin, int K, const float* W, const float* bias) {
    convert_w<<<(K * HID + 255) / 256, 256, 0, stream>>>(W, Wbf, K, HID);
    gemm_xw<<<N_NODES / 32, 256, 0, stream>>>(xin, Wbf, support, K);
    post_support<<<EW_GRID, 256, 0, stream>>>(support, inv_norm, bias, sB, xf, xbf);
    adj_side<<<N_NODES / 256, 256, 0, stream>>>(adjB, sB, bias, xf, xbf);
  };

  // one-time prep: bf16 adjacency (L2-resident thereafter) + degree normalizer
  prep_adj<<<N_NODES, 256, 0, stream>>>(adj, adjB, inv_norm);
  pack_full<<<(N_NODES * K_FULL) / 256, 256, 0, stream>>>(features, pooled, fullB);

  // gc1 (K=960) and gc2 (K=192)
  run_layer(fullB, K_FULL, W1, b1);
  run_layer(xbf, HID, Wm + 0, bm + 0);
  avg_kernel<<<EW_GRID, 256, 0, stream>>>(features, xf, feats, featsB);

  // gc3..gc12: five (A,B) pairs with residual averaging
  for (int p = 0; p < 5; ++p) {
    run_layer(featsB, HID, Wm + (size_t)(2 * p + 1) * HID * HID, bm + (2 * p + 1) * HID);
    run_layer(xbf,    HID, Wm + (size_t)(2 * p + 2) * HID * HID, bm + (2 * p + 2) * HID);
    avg_kernel<<<EW_GRID, 256, 0, stream>>>(feats, xf, feats, featsB);
  }

  // gc13 + residual
  run_layer(featsB, HID, Wm + (size_t)11 * HID * HID, bm + 11 * HID);
  avg_kernel<<<EW_GRID, 256, 0, stream>>>(feats, xf, feats, featsB);

  // gc15 (identity activation, side=2)
  coords_support<<<(N_NODES * 3 + 255) / 256, 256, 0, stream>>>(feats, W15, inv_norm, s3, sc);
  coords_side<<<N_NODES, 256, 0, stream>>>(adjB, sc, s3, b15, coords);
}
